// Node2NodeSupConLoss_23888608100754
// MI455X (gfx1250) — matrix operand order repaired and checked by reading, not compiled, not run on hardware
//
#include <hip/hip_runtime.h>

typedef __attribute__((ext_vector_type(16))) _Float16 v16h;
typedef __attribute__((ext_vector_type(8)))  _Float16 v8h;
typedef __attribute__((ext_vector_type(4)))  _Float16 v4h;
typedef __attribute__((ext_vector_type(8)))  float    v8f;

#define DIM      128      // feature dim (reference D)
#define TEMP_INV 10.0f    // 1 / TEMP

// ---------------------------------------------------------------------------
// Zero the scalar output (d_out is poisoned by the harness).
// ---------------------------------------------------------------------------
__global__ __launch_bounds__(64) void zero_out_kernel(float* out) {
    if (threadIdx.x == 0 && blockIdx.x == 0) out[0] = 0.0f;
}

// ---------------------------------------------------------------------------
// L2-normalize each 128-float row and store as f16 into workspace.
// One wave32 per row: lane loads 4 consecutive floats (128 = 32*4).
// ---------------------------------------------------------------------------
__global__ __launch_bounds__(256) void normalize_f16_kernel(
    const float* __restrict__ x, _Float16* __restrict__ xh, int nrows)
{
    const int row  = blockIdx.x * 8 + (threadIdx.x >> 5);
    const int lane = threadIdx.x & 31;
    if (row >= nrows) return;   // uniform per wave

    const float4 v = *((const float4*)(x + (size_t)row * DIM) + lane);
    float ss = v.x * v.x + v.y * v.y + v.z * v.z + v.w * v.w;
#pragma unroll
    for (int m = 16; m >= 1; m >>= 1) ss += __shfl_xor(ss, m, 32);
    const float inv = rsqrtf(ss);

    v4h h;
    h[0] = (_Float16)(v.x * inv);
    h[1] = (_Float16)(v.y * inv);
    h[2] = (_Float16)(v.z * inv);
    h[3] = (_Float16)(v.w * inv);
    *((v4h*)(xh + (size_t)row * DIM) + lane) = h;
}

// ---------------------------------------------------------------------------
// One block per anchor. 8 waves x 64 samples = 512 samples.
// Per wave: 4 tiles of 16 samples; each tile = 4 WMMA f32_16x16x32_f16
// (K = 128 in 4 chunks of 32). Anchor row is broadcast across the 16 M-rows
// of the A fragment, so C VGPR0 holds the 16 sims of the tile.
// ---------------------------------------------------------------------------
__global__ __launch_bounds__(256) void supcon_wmma_kernel(
    const _Float16* __restrict__ xh,
    const int*      __restrict__ y,
    const int*      __restrict__ anchors,
    const int*      __restrict__ sampled,
    float*          __restrict__ out,
    int S)
{
    const int a    = blockIdx.x;
    const int tid  = threadIdx.x;
    const int wave = tid >> 5;      // 0..7
    const int lane = tid & 31;
    const int sub  = lane & 15;     // sample slot within tile
    const int hi   = lane >> 4;     // half-wave select

    const int anode = anchors[a];
    const int ya    = y[anode];
    const _Float16* __restrict__ arow = xh + (size_t)anode * DIM;

    // A fragments: 16x32 f16, M x K, all M rows = anchor row.
    // ISA layout: lanes 0-15 hold K 0-7 (V0-3) and K 16-23 (V4-7);
    //             lanes 16-31 hold K 8-15 and K 24-31.
    v16h afrag[4];
#pragma unroll
    for (int k = 0; k < 4; ++k) {
        const int off = k * 32 + hi * 8;
        const v8h lo = *(const v8h*)(arow + off);
        const v8h hh = *(const v8h*)(arow + off + 16);
#pragma unroll
        for (int i = 0; i < 8; ++i) { afrag[k][i] = lo[i]; afrag[k][8 + i] = hh[i]; }
    }

    // Gather the 4 sample indices for this wave's tiles; prefetch their rows.
    const int* __restrict__ sidx_base = sampled + (size_t)a * S + wave * 64;
    int sidx[4];
#pragma unroll
    for (int t = 0; t < 4; ++t) {
        sidx[t] = sidx_base[t * 16 + sub];
        __builtin_prefetch(xh + (size_t)sidx[t] * DIM, 0, 3);
    }

    const float active = (hi == 0) ? 1.0f : 0.0f;  // lanes 16-31 duplicate data
    float num = 0.0f, den = 0.0f, cnt = 0.0f;

#pragma unroll
    for (int t = 0; t < 4; ++t) {
        const _Float16* __restrict__ srow = xh + (size_t)sidx[t] * DIM;
        v8f acc = {};
#pragma unroll
        for (int k = 0; k < 4; ++k) {
            // B fragment: 32(K) x 16(N) f16. Lanes 0-15 = sample `sub`
            // with K 0-7 and K 16-23 of the chunk; lanes 16-31 = same sample, K 8-15 and K 24-31.
            const int off = k * 32 + hi * 8;
            const v8h b0 = *(const v8h*)(srow + off);
            const v8h b1 = *(const v8h*)(srow + off + 16);
            v16h bfrag;
#pragma unroll
            for (int i = 0; i < 8; ++i) { bfrag[i] = b0[i]; bfrag[8 + i] = b1[i]; }
            acc = __builtin_amdgcn_wmma_f32_16x16x32_f16(
                /*neg_a=*/false, afrag[k], /*neg_b=*/false, bfrag,
                /*c_mod=*/(short)0, acc, /*reuse_a=*/false, /*reuse_b=*/false);
        }
        // C VGPR0, lanes 0-15: M=0 row = sim of sample (tile base + sub).
        const float sim = acc[0];
        const float e   = expf(sim * TEMP_INV);
        const float pos = (y[sidx[t]] == ya) ? 1.0f : 0.0f;
        num += active * e * pos;
        den += active * e;
        cnt += active * pos;
    }

    // Wave reduction (lanes >= 16 contribute zero).
#pragma unroll
    for (int m = 16; m >= 1; m >>= 1) {
        num += __shfl_xor(num, m, 32);
        den += __shfl_xor(den, m, 32);
        cnt += __shfl_xor(cnt, m, 32);
    }

    __shared__ float red[8][3];
    if (lane == 0) { red[wave][0] = num; red[wave][1] = den; red[wave][2] = cnt; }
    __syncthreads();

    if (tid == 0) {
        float tn = 0.0f, td = 0.0f, tc = 0.0f;
#pragma unroll
        for (int w = 0; w < 8; ++w) { tn += red[w][0]; td += red[w][1]; tc += red[w][2]; }
        float loss = 0.0f;
        if (tc > 0.0f) loss = -logf(tn / td) / tc;
        atomicAdd(out, loss);
    }
}

// ---------------------------------------------------------------------------
extern "C" void kernel_launch(void* const* d_in, const int* in_sizes, int n_in,
                              void* d_out, int out_size, void* d_ws, size_t ws_size,
                              hipStream_t stream)
{
    const float* x       = (const float*)d_in[0];   // [N, 128] f32
    const int*   y       = (const int*)  d_in[1];   // [N]
    const int*   anchors = (const int*)  d_in[2];   // [A]
    const int*   sampled = (const int*)  d_in[3];   // [A, S]
    float* out = (float*)d_out;                     // scalar

    const int N = in_sizes[1];
    const int A = in_sizes[2];
    const int S = in_sizes[3] / A;                  // 512

    _Float16* xh = (_Float16*)d_ws;                 // N * 128 f16 = 25.6 MB

    zero_out_kernel<<<1, 64, 0, stream>>>(out);
    normalize_f16_kernel<<<(N + 7) / 8, 256, 0, stream>>>(x, xh, N);
    supcon_wmma_kernel<<<A, 256, 0, stream>>>(xh, y, anchors, sampled, out, S);
}
